// FusedMoE_38697655337034
// MI455X (gfx1250) — compile-verified, matrix operand order checked by reading, not run on hardware
//
#include <hip/hip_runtime.h>
#include <cstdint>
#include <cstddef>

// ---- problem constants (match reference) ----
#define ALPHA  1.702f
#define LIMIT  7.0f
#define NTOK   1024
#define HDIM   2880
#define EDIM   16
#define IDIM   2880
#define CAP    1024          // per-expert bucket capacity (worst case)
#define ACT_OFF 262144       // byte offset of act buffer inside workspace

typedef __attribute__((ext_vector_type(16))) __bf16 v16bf;
typedef __attribute__((ext_vector_type(8)))  __bf16 v8bf;
typedef __attribute__((ext_vector_type(8)))  float  v8f;
typedef __attribute__((ext_vector_type(4)))  float  v4f;

// Load 16 bf16 WMMA-fragment elements from an fp32 row: K = [k0..k0+7] and [k0+16..k0+23]
__device__ inline v16bf ld_f32_frag(const float* __restrict__ p) {
  v4f a0 = *(const v4f*)(p);
  v4f a1 = *(const v4f*)(p + 4);
  v4f a2 = *(const v4f*)(p + 16);
  v4f a3 = *(const v4f*)(p + 20);
  v16bf r;
#pragma unroll
  for (int i = 0; i < 4; ++i) {
    r[i]      = (__bf16)a0[i];
    r[i + 4]  = (__bf16)a1[i];
    r[i + 8]  = (__bf16)a2[i];
    r[i + 12] = (__bf16)a3[i];
  }
  return r;
}

// Same, but source already bf16 (act buffer)
__device__ inline v16bf ld_bf16_frag(const __bf16* __restrict__ p) {
  v8bf lo = *(const v8bf*)(p);
  v8bf hi = *(const v8bf*)(p + 16);
  v16bf r;
#pragma unroll
  for (int i = 0; i < 8; ++i) { r[i] = lo[i]; r[i + 8] = hi[i]; }
  return r;
}

// ---------------- kernel 0: zero output + expert counters ----------------
__global__ void moe_zero(float* __restrict__ out, int n, int* __restrict__ cnt) {
  int i = blockIdx.x * blockDim.x + threadIdx.x;
  if (i < n) out[i] = 0.0f;
  if (i < EDIM) cnt[i] = 0;
}

// ---------------- kernel 1: router (logits -> top4 -> softmax -> buckets) ----------------
__global__ __launch_bounds__(256)
void moe_router(const float* __restrict__ hidden, const float* __restrict__ rw,
                const float* __restrict__ rb, int* __restrict__ cnt,
                int* __restrict__ btok, float* __restrict__ bprob) {
  const int t   = blockIdx.x;          // token
  const int tid = threadIdx.x;
  const int e     = tid & 15;          // expert
  const int chunk = tid >> 4;          // 16 chunks of 180 elements
  __shared__ float sm[256];
  __shared__ float lg[EDIM];

  const float* hrow = hidden + (size_t)t * HDIM;
  const float* wrow = rw + (size_t)e * HDIM;
  float acc = 0.0f;
  const int h0 = chunk * (HDIM / 16);
  for (int h = h0; h < h0 + (HDIM / 16); ++h) acc += hrow[h] * wrow[h];
  sm[tid] = acc;
  __syncthreads();

  if (tid < EDIM) {
    float s = 0.0f;
    for (int c = 0; c < 16; ++c) s += sm[tid + (c << 4)];
    lg[tid] = s + rb[tid];
  }
  __syncthreads();

  if (tid == 0) {
    float vals[4]; int idx[4]; unsigned used = 0;
    for (int k = 0; k < 4; ++k) {
      float best = -3.4e38f; int bi = 0;
      for (int x = 0; x < EDIM; ++x)
        if (!((used >> x) & 1u) && lg[x] > best) { best = lg[x]; bi = x; }
      used |= 1u << bi; vals[k] = best; idx[k] = bi;
    }
    float mx = vals[0], sum = 0.0f, p[4];
    for (int k = 0; k < 4; ++k) { p[k] = __expf(vals[k] - mx); sum += p[k]; }
    for (int k = 0; k < 4; ++k) {
      int ee = idx[k];
      int slot = atomicAdd(&cnt[ee], 1);
      btok[ee * CAP + slot]  = t;
      bprob[ee * CAP + slot] = p[k] / sum;
    }
  }
}

// ---------------- kernel 2: prefix sum over expert counts ----------------
__global__ void moe_prefix(const int* __restrict__ cnt, int* __restrict__ row_base) {
  if (blockIdx.x == 0 && threadIdx.x == 0) {
    int s = 0;
    for (int e = 0; e < EDIM; ++e) { row_base[e] = s; s += cnt[e]; }
  }
}

// ---------------- kernel 3: GEMM1 (hidden x w1^T) + clipped SwiGLU -> act (bf16) ----------------
// One wave per block. Tile: M = 32 tokens (2 sub-tiles), N = 16 intermediates, K loop of 32.
__global__ __launch_bounds__(32)
void moe_gemm1(const float* __restrict__ hidden, const float* __restrict__ w1,
               const float* __restrict__ b1, const int* __restrict__ cnt,
               const int* __restrict__ row_base, const int* __restrict__ btok,
               __bf16* __restrict__ act) {
  const int e = blockIdx.z;
  const int c = cnt[e];
  const int tile = blockIdx.y;                 // 32-token tile
  if (tile * 32 >= c) return;
  const int nb   = blockIdx.x * 16;            // intermediate tile base
  const int lane = threadIdx.x;
  const int lm = lane & 15, khalf = lane >> 4;

  // A rows (this lane's token for each sub-tile), clamped for tail
  int a0 = tile * 32 + lm;       if (a0 >= c) a0 = c - 1;
  int a1 = tile * 32 + 16 + lm;  if (a1 >= c) a1 = c - 1;
  const float* A0 = hidden + (size_t)btok[e * CAP + a0] * HDIM;
  const float* A1 = hidden + (size_t)btok[e * CAP + a1] * HDIM;
  // B columns: gate row 2*(nb+lm), up row 2*(nb+lm)+1 of w1[e]
  const float* Bg = w1 + ((size_t)e * (2 * IDIM) + (size_t)2 * (nb + lm)) * HDIM;
  const float* Bu = Bg + HDIM;

  v8f accG0 = {}, accU0 = {}, accG1 = {}, accU1 = {};
  const int koff = khalf * 8;
  for (int kb = 0; kb < HDIM; kb += 32) {
    const int k0 = kb + koff;
    v16bf fa0 = ld_f32_frag(A0 + k0);
    v16bf fa1 = ld_f32_frag(A1 + k0);
    v16bf fbg = ld_f32_frag(Bg + k0);
    v16bf fbu = ld_f32_frag(Bu + k0);
    accG0 = __builtin_amdgcn_wmma_f32_16x16x32_bf16(false, fa0, false, fbg, (short)0, accG0, false, false);
    accU0 = __builtin_amdgcn_wmma_f32_16x16x32_bf16(false, fa0, false, fbu, (short)0, accU0, false, false);
    accG1 = __builtin_amdgcn_wmma_f32_16x16x32_bf16(false, fa1, false, fbg, (short)0, accG1, false, false);
    accU1 = __builtin_amdgcn_wmma_f32_16x16x32_bf16(false, fa1, false, fbu, (short)0, accU1, false, false);
  }

  // epilogue: C element (m,n) = (r + 8*(lane>>4), lane&15)
  const int n   = lane & 15;
  const int mhi = lane >> 4;
  const float bg = b1[(size_t)e * 2 * IDIM + 2 * (nb + n)];
  const float bu = b1[(size_t)e * 2 * IDIM + 2 * (nb + n) + 1];
  const int rbase = row_base[e];
#pragma unroll
  for (int sub = 0; sub < 2; ++sub) {
    const v8f* ag = sub ? &accG1 : &accG0;
    const v8f* au = sub ? &accU1 : &accU0;
#pragma unroll
    for (int r = 0; r < 8; ++r) {
      int srow = tile * 32 + sub * 16 + r + 8 * mhi;
      if (srow < c) {
        float g = (*ag)[r] + bg;
        float u = (*au)[r] + bu;
        g = fminf(g, LIMIT);
        u = fminf(fmaxf(u, -LIMIT), LIMIT);
        float sg = 1.0f / (1.0f + __expf(-ALPHA * g));
        float v  = (u + 1.0f) * (g * sg);
        act[(size_t)(rbase + srow) * IDIM + (nb + n)] = (__bf16)v;
      }
    }
  }
}

// ---------------- kernel 4: GEMM2 (act x w2^T), scale by prob, atomic scatter ----------------
__global__ __launch_bounds__(32)
void moe_gemm2(const __bf16* __restrict__ act, const float* __restrict__ w2,
               const float* __restrict__ b2, const int* __restrict__ cnt,
               const int* __restrict__ row_base, const int* __restrict__ btok,
               const float* __restrict__ bprob, float* __restrict__ out) {
  const int e = blockIdx.z;
  const int c = cnt[e];
  const int tile = blockIdx.y;
  if (tile * 32 >= c) return;
  const int hb   = blockIdx.x * 16;            // output-H tile base
  const int lane = threadIdx.x;
  const int lm = lane & 15, khalf = lane >> 4;

  const int rbase = row_base[e];
  int a0 = tile * 32 + lm;       if (a0 >= c) a0 = c - 1;
  int a1 = tile * 32 + 16 + lm;  if (a1 >= c) a1 = c - 1;
  const __bf16* A0 = act + (size_t)(rbase + a0) * IDIM;
  const __bf16* A1 = act + (size_t)(rbase + a1) * IDIM;
  const float*  Bh = w2 + ((size_t)e * HDIM + (hb + lm)) * (size_t)IDIM;

  v8f acc0 = {}, acc1 = {};
  const int koff = khalf * 8;
  for (int kb = 0; kb < IDIM; kb += 32) {
    const int k0 = kb + koff;
    v16bf fa0 = ld_bf16_frag(A0 + k0);
    v16bf fa1 = ld_bf16_frag(A1 + k0);
    v16bf fb  = ld_f32_frag(Bh + k0);
    acc0 = __builtin_amdgcn_wmma_f32_16x16x32_bf16(false, fa0, false, fb, (short)0, acc0, false, false);
    acc1 = __builtin_amdgcn_wmma_f32_16x16x32_bf16(false, fa1, false, fb, (short)0, acc1, false, false);
  }

  const int n   = lane & 15;
  const int mhi = lane >> 4;
  const float b2v = b2[(size_t)e * HDIM + hb + n];
#pragma unroll
  for (int sub = 0; sub < 2; ++sub) {
    const v8f* a = sub ? &acc1 : &acc0;
#pragma unroll
    for (int r = 0; r < 8; ++r) {
      int srow = tile * 32 + sub * 16 + r + 8 * mhi;
      if (srow < c) {
        int   tok = btok[e * CAP + srow];
        float p   = bprob[e * CAP + srow];
        atomicAdd(&out[(size_t)tok * HDIM + hb + n], p * ((*a)[r] + b2v));
      }
    }
  }
}

// ---------------- launch ----------------
extern "C" void kernel_launch(void* const* d_in, const int* in_sizes, int n_in,
                              void* d_out, int out_size, void* d_ws, size_t ws_size,
                              hipStream_t stream) {
  const float* hidden = (const float*)d_in[0];
  const float* rw     = (const float*)d_in[1];
  const float* rb     = (const float*)d_in[2];
  const float* w1     = (const float*)d_in[3];
  const float* b1     = (const float*)d_in[4];
  const float* w2     = (const float*)d_in[5];
  const float* b2     = (const float*)d_in[6];
  float* out = (float*)d_out;

  char* ws = (char*)d_ws;
  int*    cnt      = (int*)(ws);                       // 16 ints
  int*    row_base = (int*)(ws + 64);                  // 16 ints
  int*    btok     = (int*)(ws + 128);                 // 16*1024 ints (64KB)
  float*  bprob    = (float*)(ws + 128 + 65536);       // 16*1024 floats (64KB)
  __bf16* actbuf   = (__bf16*)(ws + ACT_OFF);          // 4096*2880 bf16 (~23.6MB)

  const int nout = NTOK * HDIM;
  moe_zero<<<(nout + 255) / 256, 256, 0, stream>>>(out, nout, cnt);
  moe_router<<<NTOK, 256, 0, stream>>>(hidden, rw, rb, cnt, btok, bprob);
  moe_prefix<<<1, 32, 0, stream>>>(cnt, row_base);

  dim3 g1(IDIM / 16, CAP / 32, EDIM);   // (180, 32, 16)
  moe_gemm1<<<g1, 32, 0, stream>>>(hidden, w1, b1, cnt, row_base, btok, actbuf);

  dim3 g2(HDIM / 16, CAP / 32, EDIM);   // (180, 32, 16)
  moe_gemm2<<<g2, 32, 0, stream>>>(actbuf, w2, b2, cnt, row_base, btok, bprob, out);
}
